// PointMixerBlock_9242769621754
// MI455X (gfx1250) — matrix operand order repaired and checked by reading, not compiled, not run on hardware
//
#include <hip/hip_runtime.h>
#include <math.h>
#include <stdint.h>

// ---- problem constants (from reference) ----
#define NPTS 32768
#define CCH  64
#define KNB  16
#define CSH  8
#define NKF  (NPTS * KNB)        // 524288 flattened (n,k) rows
#define EPSB 1e-5f

typedef __attribute__((ext_vector_type(16))) _Float16 v16h;
typedef __attribute__((ext_vector_type(8)))  float    v8f;

// ---------------------------------------------------------------------------
// WMMA helpers (CDNA5 v_wmma_f32_16x16x32_f16, wave32)
// Fragment layouts per CDNA5 ISA 7.12.2:
//   A (16x32 f16): lane holds row M=lane%16; VGPR v pairs at
//                  k0 = (v<4?0:16) + (lane>=16?8:0) + 2*(v&3)
//   B (32x16 f16): lane holds col N=lane%16; VGPR v pairs at
//                  k0 = (v<4?0:16) + (lane>=16?8:0) + 2*(v&3)
//   C/D (16x16 f32): reg r -> (M = r + 8*(lane>=16), N = lane%16)
// ---------------------------------------------------------------------------
__device__ inline v8f wmma32(v16h a, v16h b, v8f c) {
  return __builtin_amdgcn_wmma_f32_16x16x32_f16(false, a, false, b,
                                                (short)0, c, false, false);
}

__device__ inline v8f zero8() {
  v8f z = {0.f, 0.f, 0.f, 0.f, 0.f, 0.f, 0.f, 0.f};
  return z;
}

// A fragment from one f16 row pointer (this lane's row), k window [kbase, kbase+32)
__device__ inline v16h afrag_rowptr_f16(const _Float16* rowp, int kbase) {
  int hl = (threadIdx.x >> 4) & 1;
  v16h a;
#pragma unroll
  for (int v = 0; v < 8; ++v) {
    int k0 = kbase + ((v & 4) << 2) + (hl << 3) + 2 * (v & 3);
    a[2 * v]     = rowp[k0];
    a[2 * v + 1] = rowp[k0 + 1];
  }
  return a;
}

// A fragment from an LDS fp32 tile base[16][ld], zero-padded beyond kmax
__device__ inline v16h afrag_lds(const float* base, int ld, int kbase, int kmax) {
  int lane = threadIdx.x & 31;
  int m = lane & 15, hl = lane >> 4;
  v16h a;
#pragma unroll
  for (int v = 0; v < 8; ++v) {
    int k0 = ((v & 4) << 2) + (hl << 3) + 2 * (v & 3);
    int k = kbase + k0;
    a[2 * v]     = (k < kmax)     ? (_Float16)base[m * ld + k]     : (_Float16)0.0f;
    a[2 * v + 1] = (k + 1 < kmax) ? (_Float16)base[m * ld + k + 1] : (_Float16)0.0f;
  }
  return a;
}

// B fragment from global fp32 weight matrix B[k][n] (row-major, leading dim ld),
// zero-padded beyond kmax rows / nmax cols. (used once, in the pack kernel)
__device__ inline v16h bfrag_f32(const float* B, int ld, int kbase, int nbase,
                                 int kmax, int nmax) {
  int lane = threadIdx.x & 31;
  int nn = lane & 15, hl = lane >> 4;
  bool okn = nn < nmax;
  v16h b;
#pragma unroll
  for (int v = 0; v < 8; ++v) {
    int k = kbase + ((v & 4) << 2) + (hl << 3) + 2 * (v & 3);
    b[2 * v]     = (okn && k < kmax)
                     ? (_Float16)B[(size_t)k * ld + nbase + nn] : (_Float16)0.0f;
    b[2 * v + 1] = (okn && k + 1 < kmax)
                     ? (_Float16)B[(size_t)(k + 1) * ld + nbase + nn] : (_Float16)0.0f;
  }
  return b;
}

// pre-packed B fragment fetch: contiguous 32B per lane -> global_load_b128 x2
__device__ inline v16h bfrag_pk(const v16h* bf, int frag) {
  return bf[frag * 32 + (threadIdx.x & 31)];
}

// fragment table bases inside the packed buffer
#define BF_W1   0   // 8: [ks*4+ct]
#define BF_M3W  8   // 8
#define BF_W3   16  // 8
#define BF_M1W1 24  // 2: [ks]
#define BF_M1W2 26  // 1
#define BF_M2W1 27  // 4: [ct]
#define BF_M2W2 31  // 2: [ks]
#define BF_NUM  33

// order-preserving float->uint encoding for atomic max on floats
__device__ inline unsigned fenc(float f) {
  unsigned u = __float_as_uint(f);
  return (u >> 31) ? ~u : (u | 0x80000000u);
}
__device__ inline float fdec(unsigned u) {
  return __uint_as_float((u >> 31) ? (u & 0x7fffffffu) : ~u);
}

// ---------------------------------------------------------------------------
// k_zero: clear stats + segment buffers (must happen every launch)
// ---------------------------------------------------------------------------
__global__ void pmx_zero(float* p, int n) {
  int t = blockIdx.x * 256 + threadIdx.x;
  if (t < n) p[t] = 0.f;
}

// fp32 -> f16 bulk convert
__global__ void pmx_cvt(const float* __restrict__ src, _Float16* __restrict__ dst) {
  int t = blockIdx.x * 256 + threadIdx.x;
  dst[t] = (_Float16)src[t];
}

// ---------------------------------------------------------------------------
// One-time pack of all WMMA B fragments into the exact lane/VGPR layout.
// One 32-thread block per fragment.
// ---------------------------------------------------------------------------
__global__ void __launch_bounds__(32) pmx_pack(
    const float* __restrict__ w1, const float* __restrict__ m3w,
    const float* __restrict__ w3, const float* __restrict__ m1w1,
    const float* __restrict__ m1w2, const float* __restrict__ m2w1,
    const float* __restrict__ m2w2, v16h* __restrict__ bf) {
  int b = blockIdx.x;
  v16h v;
  if (b < 8) {
    v = bfrag_f32(w1, 64, (b >> 2) * 32, (b & 3) * 16, 64, 16);
  } else if (b < 16) {
    int i = b - 8;
    v = bfrag_f32(m3w, 64, (i >> 2) * 32, (i & 3) * 16, 64, 16);
  } else if (b < 24) {
    int i = b - 16;
    v = bfrag_f32(w3, 64, (i >> 2) * 32, (i & 3) * 16, 64, 16);
  } else if (b < 26) {
    int i = b - 24;
    v = bfrag_f32(m1w1 + 3 * 16, 16, i * 32, 0, 64, 16);  // rows 3..66 of m1w1
  } else if (b < 27) {
    v = bfrag_f32(m1w2, 16, 0, 0, 16, 16);                 // K padded 16->32
  } else if (b < 31) {
    int i = b - 27;
    v = bfrag_f32(m2w1, 64, 0, i * 16, 32, 16);
  } else {
    int i = b - 31;
    v = bfrag_f32(m2w2, 8, i * 32, 0, 64, 8);              // N padded 8->16
  }
  bf[b * 32 + threadIdx.x] = v;
}

// ---------------------------------------------------------------------------
// Generic (Nx64)@(64x64)+bias GEMM, one wave per 16-row tile, WMMA f16,
// A pre-converted to f16, B from packed fragments.
// Optional per-column sum / sumsq accumulation (LDS-staged, then global atomics)
// ---------------------------------------------------------------------------
__global__ void __launch_bounds__(256) pmx_gemm64(
    const _Float16* __restrict__ A, const v16h* __restrict__ bf, int bfbase,
    const float* __restrict__ bias, float* __restrict__ Y,
    float* __restrict__ ssum, float* __restrict__ ssq) {
  __shared__ float st[128];
  if (threadIdx.x < 128) st[threadIdx.x] = 0.f;
  __syncthreads();

  int w = threadIdx.x >> 5, lane = threadIdx.x & 31;
  int m = lane & 15, hl = lane >> 4;
  int rt = blockIdx.x * 8 + w;  // 16-row tile index

  const _Float16* arow = A + ((size_t)rt * 16 + m) * CCH;
  v16h a0 = afrag_rowptr_f16(arow, 0);
  v16h a1 = afrag_rowptr_f16(arow, 32);

  for (int ct = 0; ct < 4; ++ct) {
    v8f z = zero8();
    z = wmma32(a0, bfrag_pk(bf, bfbase + ct), z);
    z = wmma32(a1, bfrag_pk(bf, bfbase + 4 + ct), z);
    int c = ct * 16 + m;
    float bi = bias ? bias[c] : 0.f;
    float ls = 0.f, lq = 0.f;
#pragma unroll
    for (int r = 0; r < 8; ++r) {
      float v = z[r] + bi;
      Y[((size_t)rt * 16 + r + 8 * hl) * CCH + c] = v;
      ls += v;
      lq += v * v;
    }
    if (ssum) {
      atomicAdd(&st[c], ls);
      atomicAdd(&st[64 + c], lq);
    }
  }
  __syncthreads();
  if (ssum && threadIdx.x < 64) {
    atomicAdd(&ssum[threadIdx.x], st[threadIdx.x]);
    atomicAdd(&ssq[threadIdx.x], st[64 + threadIdx.x]);
  }
}

// ---------------------------------------------------------------------------
// BN(3) stats for linear_p: q = p_r @ lpw1 + lpb1 over all (n,k)
// ---------------------------------------------------------------------------
__global__ void __launch_bounds__(256) pmx_lp_stats(
    const float* __restrict__ p, const int* __restrict__ knn,
    const float* __restrict__ lpw1, const float* __restrict__ lpb1,
    float* __restrict__ lp_sum, float* __restrict__ lp_sq) {
  __shared__ float red[6];
  if (threadIdx.x < 6) red[threadIdx.x] = 0.f;
  __syncthreads();
  int tid = blockIdx.x * 256 + threadIdx.x;
  int n = tid >> 4;
  int id = knn[tid];
  float pr0 = p[(size_t)id * 3 + 0] - p[(size_t)n * 3 + 0];
  float pr1 = p[(size_t)id * 3 + 1] - p[(size_t)n * 3 + 1];
  float pr2 = p[(size_t)id * 3 + 2] - p[(size_t)n * 3 + 2];
#pragma unroll
  for (int d = 0; d < 3; ++d) {
    float q = lpb1[d] + pr0 * lpw1[d] + pr1 * lpw1[3 + d] + pr2 * lpw1[6 + d];
    atomicAdd(&red[d], q);
    atomicAdd(&red[3 + d], q * q);
  }
  __syncthreads();
  if (threadIdx.x < 3) {
    atomicAdd(&lp_sum[threadIdx.x], red[threadIdx.x]);
    atomicAdd(&lp_sq[threadIdx.x], red[3 + threadIdx.x]);
  }
}

// ---------------------------------------------------------------------------
// Elementwise BN(train) + ReLU over (N, 64), channel = last index
// ---------------------------------------------------------------------------
__global__ void pmx_bn_relu(const float* __restrict__ src, float* __restrict__ dst,
                            const float* __restrict__ sum, const float* __restrict__ sq,
                            const float* __restrict__ g, const float* __restrict__ b,
                            float cinv) {
  int tid = blockIdx.x * 256 + threadIdx.x;
  int c = tid & 63;
  float mean = sum[c] * cinv;
  float var = sq[c] * cinv - mean * mean;
  float v = (src[tid] - mean) * rsqrtf(var + EPSB) * g[c] + b[c];
  dst[tid] = fmaxf(v, 0.f);
}

// tail: relu(bn3(t3) + identity)
__global__ void pmx_bn3_out(const float* __restrict__ t3, const float* __restrict__ xin,
                            const float* __restrict__ sum, const float* __restrict__ sq,
                            const float* __restrict__ g, const float* __restrict__ b,
                            float cinv, float* __restrict__ out) {
  int tid = blockIdx.x * 256 + threadIdx.x;
  int c = tid & 63;
  float mean = sum[c] * cinv;
  float var = sq[c] * cinv - mean * mean;
  float v = (t3[tid] - mean) * rsqrtf(var + EPSB) * g[c] + b[c] + xin[tid];
  out[tid] = fmaxf(v, 0.f);
}

// ---------------------------------------------------------------------------
// Intra pass 1: one wave per point.
//   energy = relu(x_knn @ m1w1 + m1b1) @ m1w2 + m1b2            (WMMA x3)
//   pe     = relu(bn(p_r @ lpw1 + lpb1)) @ lpw2 + lpb2          (VALU)
//   e      = [energy | shrink(pe)]  -> stored f16
//   stats of e @ m2w1 (pre-BN) accumulated                      (WMMA x4)
// ---------------------------------------------------------------------------
__global__ void __launch_bounds__(256) pmx_intra1(
    const float* __restrict__ p, const int* __restrict__ knn,
    const _Float16* __restrict__ h16, const v16h* __restrict__ bf,
    const float* __restrict__ m1w1, const float* __restrict__ m1b1,
    const float* __restrict__ m1b2,
    const float* __restrict__ lpw1, const float* __restrict__ lpb1,
    const float* __restrict__ lpg, const float* __restrict__ lpbt,
    const float* __restrict__ lpw2, const float* __restrict__ lpb2,
    const float* __restrict__ lp_sum, const float* __restrict__ lp_sq,
    _Float16* __restrict__ e_ws, _Float16* __restrict__ pe_ws,
    float* __restrict__ e1_sum, float* __restrict__ e1_sq) {
  __shared__ float s_pr[8][16][3];
  __shared__ float s_qn[8][16][3];
  __shared__ float s_pe[8][16][64];
  __shared__ float s_e[8][16][32];
  __shared__ int   s_idx[8][16];
  __shared__ float s_stat[128];

  int w = threadIdx.x >> 5, lane = threadIdx.x & 31;
  int m = lane & 15, hl = lane >> 4;
  int n = blockIdx.x * 8 + w;

  if (threadIdx.x < 128) s_stat[threadIdx.x] = 0.f;
  if (lane < 16) {
    int id = knn[n * KNB + lane];
    s_idx[w][lane] = id;
#pragma unroll
    for (int d = 0; d < 3; ++d)
      s_pr[w][lane][d] = p[(size_t)id * 3 + d] - p[(size_t)n * 3 + d];
  }
  __syncthreads();

  // ---- energy stage 1: hknn(16x64) @ m1w1[3:,:](64x16) ----
  const _Float16* rowp = h16 + (size_t)s_idx[w][m] * CCH;
  v8f acc = zero8();
  acc = wmma32(afrag_rowptr_f16(rowp, 0),  bfrag_pk(bf, BF_M1W1 + 0), acc);
  acc = wmma32(afrag_rowptr_f16(rowp, 32), bfrag_pk(bf, BF_M1W1 + 1), acc);
#pragma unroll
  for (int r = 0; r < 8; ++r) {
    int M = r + 8 * hl;
    float v = m1b1[m];
#pragma unroll
    for (int d = 0; d < 3; ++d) v += s_pr[w][M][d] * m1w1[d * 16 + m];
    s_e[w][M][m] = fmaxf(acc[r] + v, 0.f);  // relu, staged as A for next WMMA
  }
  __syncthreads();

  // ---- energy stage 2: (16x16, K padded to 32) @ m1w2(16x16) ----
  v8f acc2 = zero8();
  acc2 = wmma32(afrag_lds(&s_e[w][0][0], 32, 0, 16), bfrag_pk(bf, BF_M1W2), acc2);
  __syncthreads();
#pragma unroll
  for (int r = 0; r < 8; ++r)
    s_e[w][r + 8 * hl][m] = acc2[r] + m1b2[m];  // energy -> e cols 0..15

  // ---- pe path ----
  float cinvlp = 1.f / (float)NKF;
  if (lane < 16) {
#pragma unroll
    for (int d = 0; d < 3; ++d) {
      float q = lpb1[d];
#pragma unroll
      for (int t = 0; t < 3; ++t) q += s_pr[w][lane][t] * lpw1[t * 3 + d];
      float mean = lp_sum[d] * cinvlp;
      float var = lp_sq[d] * cinvlp - mean * mean;
      s_qn[w][lane][d] =
          fmaxf((q - mean) * rsqrtf(var + EPSB) * lpg[d] + lpbt[d], 0.f);
    }
  }
  __syncthreads();
  for (int c = lane; c < 64; c += 32) {
#pragma unroll
    for (int j = 0; j < 16; ++j) {
      float v = lpb2[c] + s_qn[w][j][0] * lpw2[c] + s_qn[w][j][1] * lpw2[64 + c] +
                s_qn[w][j][2] * lpw2[128 + c];
      s_pe[w][j][c] = v;
      pe_ws[((size_t)n * KNB + j) * CCH + c] = (_Float16)v;
    }
  }
  __syncthreads();
  // pe_shrink -> e cols 16..31
  for (int i = lane; i < 256; i += 32) {
    int j = i >> 4, t = i & 15;
    s_e[w][j][16 + t] =
        s_pe[w][j][t] + s_pe[w][j][16 + t] + s_pe[w][j][32 + t] + s_pe[w][j][48 + t];
  }
  __syncthreads();
  // persist e (f16) for passes 2/3
  for (int i = lane; i < 512; i += 32) {
    int j = i >> 5, t = i & 31;
    e_ws[((size_t)n * KNB + j) * 32 + t] = (_Float16)s_e[w][j][t];
  }

  // ---- stats of e1_pre = e(16x32) @ m2w1(32x64) ----
  v16h ea = afrag_lds(&s_e[w][0][0], 32, 0, 32);
  for (int ct = 0; ct < 4; ++ct) {
    v8f z = zero8();
    z = wmma32(ea, bfrag_pk(bf, BF_M2W1 + ct), z);
    float ls = 0.f, lq = 0.f;
#pragma unroll
    for (int r = 0; r < 8; ++r) {
      ls += z[r];
      lq += z[r] * z[r];
    }
    atomicAdd(&s_stat[ct * 16 + m], ls);
    atomicAdd(&s_stat[64 + ct * 16 + m], lq);
  }
  __syncthreads();
  if (threadIdx.x < 64) {
    atomicAdd(&e1_sum[threadIdx.x], s_stat[threadIdx.x]);
    atomicAdd(&e1_sq[threadIdx.x], s_stat[64 + threadIdx.x]);
  }
}

// ---------------------------------------------------------------------------
// Intra pass 2: replay e@m2w1 from stored e, BN+ReLU, stats of e1 @ m2w2
// ---------------------------------------------------------------------------
__global__ void __launch_bounds__(256) pmx_intra2(
    const _Float16* __restrict__ e_ws, const v16h* __restrict__ bf,
    const float* __restrict__ m2g1, const float* __restrict__ m2bt1,
    const float* __restrict__ e1_sum, const float* __restrict__ e1_sq,
    float* __restrict__ e2_sum, float* __restrict__ e2_sq) {
  __shared__ float s_e1[8][16][64];
  __shared__ float st[16];
  int w = threadIdx.x >> 5, lane = threadIdx.x & 31;
  int m = lane & 15, hl = lane >> 4;
  int n = blockIdx.x * 8 + w;
  if (threadIdx.x < 16) st[threadIdx.x] = 0.f;
  __syncthreads();

  float cinv = 1.f / (float)NKF;
  v16h ea = afrag_rowptr_f16(e_ws + ((size_t)n * KNB + m) * 32, 0);
  for (int ct = 0; ct < 4; ++ct) {
    v8f z = zero8();
    z = wmma32(ea, bfrag_pk(bf, BF_M2W1 + ct), z);
    int c = ct * 16 + m;
    float mean = e1_sum[c] * cinv;
    float var = e1_sq[c] * cinv - mean * mean;
    float sc = rsqrtf(var + EPSB) * m2g1[c], sh = m2bt1[c];
#pragma unroll
    for (int r = 0; r < 8; ++r)
      s_e1[w][r + 8 * hl][c] = fmaxf((z[r] - mean) * sc + sh, 0.f);
  }
  __syncthreads();

  v8f z2 = zero8();
  z2 = wmma32(afrag_lds(&s_e1[w][0][0], 64, 0, 64),  bfrag_pk(bf, BF_M2W2 + 0), z2);
  z2 = wmma32(afrag_lds(&s_e1[w][0][0], 64, 32, 64), bfrag_pk(bf, BF_M2W2 + 1), z2);
  if (m < 8) {
    float ls = 0.f, lq = 0.f;
#pragma unroll
    for (int r = 0; r < 8; ++r) {
      ls += z2[r];
      lq += z2[r] * z2[r];
    }
    atomicAdd(&st[m], ls);
    atomicAdd(&st[8 + m], lq);
  }
  __syncthreads();
  if (threadIdx.x < 8) {
    atomicAdd(&e2_sum[threadIdx.x], st[threadIdx.x]);
    atomicAdd(&e2_sq[threadIdx.x], st[8 + threadIdx.x]);
  }
}

// ---------------------------------------------------------------------------
// Intra pass 3: finish — softmax weights, xk, x_intra, s, scatter-max.
// The wave's contiguous 2KB pe tile is brought into LDS with
// GLOBAL_LOAD_ASYNC_TO_LDS_B128 (ASYNCcnt path), then consumed from LDS.
// ---------------------------------------------------------------------------
__global__ void __launch_bounds__(256) pmx_intra3(
    const int* __restrict__ knn, const _Float16* __restrict__ e_ws,
    const _Float16* __restrict__ pe_ws, const float* __restrict__ xv,
    const v16h* __restrict__ bf,
    const float* __restrict__ m2g1, const float* __restrict__ m2bt1,
    const float* __restrict__ m2g2, const float* __restrict__ m2bt2,
    const float* __restrict__ m2w3, const float* __restrict__ m2b3,
    const float* __restrict__ iw, const float* __restrict__ ib,
    const float* __restrict__ e1_sum, const float* __restrict__ e1_sq,
    const float* __restrict__ e2_sum, const float* __restrict__ e2_sq,
    _Float16* __restrict__ xk_ws, float* __restrict__ s_out,
    float* __restrict__ x_intra, unsigned* __restrict__ smax_enc) {
  __shared__ float    s_e1[8][16][64];  // reused later as xk staging
  __shared__ _Float16 s_pe16[8][16][64];
  __shared__ float    s_e2[8][16][8];
  __shared__ float    s_w[8][16][8];
  __shared__ int      s_idx[8][16];

  int w = threadIdx.x >> 5, lane = threadIdx.x & 31;
  int m = lane & 15, hl = lane >> 4;
  int n = blockIdx.x * 8 + w;

  // ---- async copy of this wave's pe tile (16x64 f16 = 2KB) into LDS ----
  {
    unsigned ldsbase = (unsigned)(uintptr_t)(void*)&s_pe16[w][0][0];
    const _Float16* gsrc = pe_ws + (size_t)n * KNB * CCH;
#pragma unroll
    for (int it = 0; it < 4; ++it) {
      unsigned ldsa = ldsbase + (unsigned)((it * 32 + lane) * 16);
      unsigned long long ga =
          (unsigned long long)(uintptr_t)gsrc + (unsigned long long)((it * 32 + lane) * 16);
      asm volatile("global_load_async_to_lds_b128 %0, %1, off"
                   :: "v"(ldsa), "v"(ga) : "memory");
    }
  }

  if (lane < 16) {
    int id = knn[n * KNB + lane];
    s_idx[w][lane] = id;
    __builtin_prefetch(xv + (size_t)id * CCH, 0, 3);  // global_prefetch_b8
  }
  __syncthreads();

  float cinv = 1.f / (float)NKF;
  // e1 = relu(bn(e @ m2w1))
  v16h ea = afrag_rowptr_f16(e_ws + ((size_t)n * KNB + m) * 32, 0);
  for (int ct = 0; ct < 4; ++ct) {
    v8f z = zero8();
    z = wmma32(ea, bfrag_pk(bf, BF_M2W1 + ct), z);
    int c = ct * 16 + m;
    float mean = e1_sum[c] * cinv;
    float var = e1_sq[c] * cinv - mean * mean;
    float sc = rsqrtf(var + EPSB) * m2g1[c], sh = m2bt1[c];
#pragma unroll
    for (int r = 0; r < 8; ++r)
      s_e1[w][r + 8 * hl][c] = fmaxf((z[r] - mean) * sc + sh, 0.f);
  }
  __syncthreads();

  // e2 = relu(bn(e1 @ m2w2))
  v8f z2 = zero8();
  z2 = wmma32(afrag_lds(&s_e1[w][0][0], 64, 0, 64),  bfrag_pk(bf, BF_M2W2 + 0), z2);
  z2 = wmma32(afrag_lds(&s_e1[w][0][0], 64, 32, 64), bfrag_pk(bf, BF_M2W2 + 1), z2);
  if (m < 8) {
    float mean = e2_sum[m] * cinv;
    float var = e2_sq[m] * cinv - mean * mean;
    float sc = rsqrtf(var + EPSB) * m2g2[m], sh = m2bt2[m];
#pragma unroll
    for (int r = 0; r < 8; ++r)
      s_e2[w][r + 8 * hl][m] = fmaxf((z2[r] - mean) * sc + sh, 0.f);
  }
  __syncthreads();

  // wpre = e2 @ m2w3 + m2b3 (16x8)
  for (int i = lane; i < 128; i += 32) {
    int j = i >> 3, q = i & 7;
    float v = m2b3[q];
#pragma unroll
    for (int t = 0; t < 8; ++t) v += s_e2[w][j][t] * m2w3[t * 8 + q];
    s_w[w][j][q] = v;
  }
  __syncthreads();
  // softmax over the k axis (16 rows) per column q
  if (lane < 8) {
    int q = lane;
    float mx = -3.0e38f;
#pragma unroll
    for (int j = 0; j < 16; ++j) mx = fmaxf(mx, s_w[w][j][q]);
    float sum = 0.f;
#pragma unroll
    for (int j = 0; j < 16; ++j) {
      float e = __expf(s_w[w][j][q] - mx);
      s_w[w][j][q] = e;
      sum += e;
    }
    float inv = 1.f / sum;
#pragma unroll
    for (int j = 0; j < 16; ++j) s_w[w][j][q] *= inv;
  }
  // make sure the async pe tile has landed before consuming it
  asm volatile("s_wait_asynccnt 0" ::: "memory");
  __syncthreads();

  // xk = (xv[idx] + pe) * w[..., c%8]; x_intra = sum_k xk
  for (int c = lane; c < 64; c += 32) {
    float acc = 0.f;
#pragma unroll
    for (int j = 0; j < 16; ++j) {
      int id = s_idx[w][j];
      float xkv = (xv[(size_t)id * CCH + c] + (float)s_pe16[w][j][c]) *
                  s_w[w][j][c & 7];
      s_e1[w][j][c] = xkv;  // stage for the s-dot below
      xk_ws[((size_t)n * KNB + j) * CCH + c] = (_Float16)xkv;
      acc += xkv;
    }
    x_intra[(size_t)n * CCH + c] = acc;
  }
  __syncthreads();

  // s = xk @ iw + ib; scatter-max into segment idx (knn value)
  for (int i = lane; i < 128; i += 32) {
    int j = i >> 3, q = i & 7;
    float sv = ib[q];
#pragma unroll
    for (int t = 0; t < CCH; ++t) sv += s_e1[w][j][t] * iw[t * 8 + q];
    s_out[((size_t)n * KNB + j) * 8 + q] = sv;
    atomicMax(&smax_enc[(size_t)s_idx[w][j] * 8 + q], fenc(sv));
  }
}

// ---------------------------------------------------------------------------
// Inter pass 2: ex = exp(s - smax[seg]); denom[seg] += ex
// ---------------------------------------------------------------------------
__global__ void pmx_inter2(const int* __restrict__ knn, const float* __restrict__ s,
                           const unsigned* __restrict__ smax_enc,
                           float* __restrict__ ex, float* __restrict__ denom) {
  int tid = blockIdx.x * 256 + threadIdx.x;  // over NKF*8
  int row = tid >> 3, q = tid & 7;
  int seg = knn[row];
  float e = __expf(s[tid] - fdec(smax_enc[(size_t)seg * 8 + q]));
  ex[tid] = e;
  atomicAdd(&denom[(size_t)seg * 8 + q], e);
}

// ---------------------------------------------------------------------------
// Inter pass 3: residual[seg] += (xk @ ixw + ixb) * (ex / denom[seg])
// ---------------------------------------------------------------------------
__global__ void pmx_inter3(const int* __restrict__ knn,
                           const _Float16* __restrict__ xk_ws,
                           const float* __restrict__ ixw, const float* __restrict__ ixb,
                           const float* __restrict__ ex, const float* __restrict__ denom,
                           float* __restrict__ resid) {
  int row = blockIdx.x * 256 + threadIdx.x;  // over NKF
  int seg = knn[row];
  const _Float16* xr = xk_ws + (size_t)row * CCH;
  float acc[8];
#pragma unroll
  for (int q = 0; q < 8; ++q) acc[q] = ixb[q];
  for (int c = 0; c < CCH; ++c) {
    float xvv = (float)xr[c];
#pragma unroll
    for (int q = 0; q < 8; ++q) acc[q] += xvv * ixw[c * 8 + q];
  }
#pragma unroll
  for (int q = 0; q < 8; ++q) {
    float prob = ex[(size_t)row * 8 + q] / denom[(size_t)seg * 8 + q];
    atomicAdd(&resid[(size_t)seg * 8 + q], acc[q] * prob);
  }
}

// ---------------------------------------------------------------------------
// x2 = x_intra + tile(residual, SHARE), plus bn2 batch stats
// ---------------------------------------------------------------------------
__global__ void pmx_x2stats(const float* __restrict__ xi, const float* __restrict__ resid,
                            float* __restrict__ x2, float* __restrict__ ssum,
                            float* __restrict__ ssq) {
  __shared__ float st[128];
  if (threadIdx.x < 128) st[threadIdx.x] = 0.f;
  __syncthreads();
  int tid = blockIdx.x * 256 + threadIdx.x;
  int n = tid >> 6, c = tid & 63;
  float v = xi[tid] + resid[(size_t)n * 8 + (c & 7)];
  x2[tid] = v;
  atomicAdd(&st[c], v);
  atomicAdd(&st[64 + c], v * v);
  __syncthreads();
  if (threadIdx.x < 64) {
    atomicAdd(&ssum[threadIdx.x], st[threadIdx.x]);
    atomicAdd(&ssq[threadIdx.x], st[64 + threadIdx.x]);
  }
}

// ---------------------------------------------------------------------------
// host launcher
// ---------------------------------------------------------------------------
extern "C" void kernel_launch(void* const* d_in, const int* in_sizes, int n_in,
                              void* d_out, int out_size, void* d_ws, size_t ws_size,
                              hipStream_t stream) {
  (void)in_sizes; (void)n_in; (void)out_size; (void)ws_size;
  const float* p    = (const float*)d_in[0];
  const float* x    = (const float*)d_in[1];
  const int*   knn  = (const int*)d_in[2];
  const float* w1   = (const float*)d_in[3];
  const float* g1   = (const float*)d_in[4];
  const float* b1   = (const float*)d_in[5];
  const float* m1w1 = (const float*)d_in[6];
  const float* m1b1 = (const float*)d_in[7];
  const float* m1w2 = (const float*)d_in[8];
  const float* m1b2 = (const float*)d_in[9];
  const float* lpw1 = (const float*)d_in[10];
  const float* lpb1 = (const float*)d_in[11];
  const float* lpg  = (const float*)d_in[12];
  const float* lpbt = (const float*)d_in[13];
  const float* lpw2 = (const float*)d_in[14];
  const float* lpb2 = (const float*)d_in[15];
  const float* m2w1 = (const float*)d_in[16];
  const float* m2g1 = (const float*)d_in[17];
  const float* m2bt1= (const float*)d_in[18];
  const float* m2w2 = (const float*)d_in[19];
  const float* m2g2 = (const float*)d_in[20];
  const float* m2bt2= (const float*)d_in[21];
  const float* m2w3 = (const float*)d_in[22];
  const float* m2b3 = (const float*)d_in[23];
  const float* m3w  = (const float*)d_in[24];
  const float* m3b  = (const float*)d_in[25];
  const float* iw   = (const float*)d_in[26];
  const float* ib   = (const float*)d_in[27];
  const float* ixw  = (const float*)d_in[28];
  const float* ixb  = (const float*)d_in[29];
  const float* g2   = (const float*)d_in[30];
  const float* b2   = (const float*)d_in[31];
  const float* w3   = (const float*)d_in[32];
  const float* g3   = (const float*)d_in[33];
  const float* b3   = (const float*)d_in[34];
  float* out = (float*)d_out;

  char* wsb = (char*)d_ws;
  auto F = [&](size_t mb) { return (float*)(wsb + mb * 1024ull * 1024ull); };

  // stats region (zeroed each launch)
  float* STATS   = F(0);
  float* lp_sum  = STATS + 0;
  float* lp_sq   = STATS + 4;
  float* e1_sum  = STATS + 8;
  float* e1_sq   = STATS + 72;
  float* e2_sum  = STATS + 136;
  float* e2_sq   = STATS + 144;
  float* bn1_sum = STATS + 152;
  float* bn1_sq  = STATS + 216;
  float* x2_sum  = STATS + 280;
  float* x2_sq   = STATS + 344;
  float* bn3_sum = STATS + 408;
  float* bn3_sq  = STATS + 472;
  unsigned* smax = (unsigned*)(STATS + 1024);
  float* denom   = STATS + 1024 + NPTS * 8;
  float* resid   = denom + NPTS * 8;
  const int ZERO_FLOATS = 1024 + 3 * NPTS * 8;

  // packed B fragments + large buffers
  v16h* BFRG = (v16h*)F(4);
  float* T1 = F(8);
  float* H  = F(16);
  float* XV = F(24);
  float* XI = F(32);
  float* X2 = F(40);
  float* Y2 = F(48);
  float* T3 = F(56);
  float* S  = F(64);
  float* EX = F(80);
  _Float16* X16  = (_Float16*)F(96);
  _Float16* H16  = (_Float16*)F(100);
  _Float16* Y216 = (_Float16*)F(104);
  _Float16* E16  = (_Float16*)(wsb + 112ull * 1024 * 1024);
  _Float16* PE16 = (_Float16*)(wsb + 144ull * 1024 * 1024);
  _Float16* XK16 = (_Float16*)(wsb + 208ull * 1024 * 1024);

  const float invN = 1.f / (float)NPTS;
  const int ELEMS = NPTS * CCH;

  pmx_zero<<<(ZERO_FLOATS + 255) / 256, 256, 0, stream>>>(STATS, ZERO_FLOATS);
  pmx_pack<<<BF_NUM, 32, 0, stream>>>(w1, m3w, w3, m1w1, m1w2, m2w1, m2w2, BFRG);
  pmx_cvt<<<ELEMS / 256, 256, 0, stream>>>(x, X16);

  // t1 = x @ w1 (+ bn1 stats)
  pmx_gemm64<<<NPTS / 128, 256, 0, stream>>>(X16, BFRG, BF_W1, nullptr, T1,
                                             bn1_sum, bn1_sq);
  // linear_p BN stats
  pmx_lp_stats<<<NKF / 256, 256, 0, stream>>>(p, knn, lpw1, lpb1, lp_sum, lp_sq);
  // h = relu(bn1(t1))
  pmx_bn_relu<<<ELEMS / 256, 256, 0, stream>>>(T1, H, bn1_sum, bn1_sq, g1, b1, invN);
  pmx_cvt<<<ELEMS / 256, 256, 0, stream>>>(H, H16);
  // xv = h @ m3w + m3b
  pmx_gemm64<<<NPTS / 128, 256, 0, stream>>>(H16, BFRG, BF_M3W, m3b, XV,
                                             nullptr, nullptr);

  // intra set layer (3 passes, one wave per point)
  pmx_intra1<<<NPTS / 8, 256, 0, stream>>>(p, knn, H16, BFRG, m1w1, m1b1, m1b2,
                                           lpw1, lpb1, lpg, lpbt, lpw2, lpb2,
                                           lp_sum, lp_sq, E16, PE16, e1_sum, e1_sq);
  pmx_intra2<<<NPTS / 8, 256, 0, stream>>>(E16, BFRG, m2g1, m2bt1,
                                           e1_sum, e1_sq, e2_sum, e2_sq);
  pmx_intra3<<<NPTS / 8, 256, 0, stream>>>(knn, E16, PE16, XV, BFRG,
                                           m2g1, m2bt1, m2g2, m2bt2, m2w3, m2b3,
                                           iw, ib, e1_sum, e1_sq, e2_sum, e2_sq,
                                           XK16, S, XI, smax);

  // inter set layer (scatter softmax)
  pmx_inter2<<<NKF * 8 / 256, 256, 0, stream>>>(knn, S, smax, EX, denom);
  pmx_inter3<<<NKF / 256, 256, 0, stream>>>(knn, XK16, ixw, ixb, EX, denom, resid);

  // tail
  pmx_x2stats<<<ELEMS / 256, 256, 0, stream>>>(XI, resid, X2, x2_sum, x2_sq);
  pmx_bn_relu<<<ELEMS / 256, 256, 0, stream>>>(X2, Y2, x2_sum, x2_sq, g2, b2, invN);
  pmx_cvt<<<ELEMS / 256, 256, 0, stream>>>(Y2, Y216);
  pmx_gemm64<<<NPTS / 128, 256, 0, stream>>>(Y216, BFRG, BF_W3, nullptr, T3,
                                             bn3_sum, bn3_sq);
  pmx_bn3_out<<<ELEMS / 256, 256, 0, stream>>>(T3, x, bn3_sum, bn3_sq, g3, b3,
                                               invN, out);
}
